// CSAM_62706522521791
// MI455X (gfx1250) — matrix operand order repaired and checked by reading, not compiled, not run on hardware
//
#include <hip/hip_runtime.h>
#include <hip/hip_bf16.h>

// ---------------------------------------------------------------------------
// CDNA5 (gfx1250) implementation of the cross-scale attention module (CSAM).
// All GEMM work (conv1x1, QKV/out projections, attention score & context
// matmuls) runs on the WMMA matrix pipe: fp32 operands converted to f16,
// v_wmma_f32_16x16x32_f16 with fp32 accumulation. B tiles staged through LDS
// as f16 in [n][k] layout and read back as ds_load_b128; A fragments loaded
// as global_load_b128. Full K-chunks are completely unmasked; only the K%32
// tail (K=400 in the ctx GEMM) takes a masked path once.
// ---------------------------------------------------------------------------

typedef __attribute__((ext_vector_type(16))) _Float16 v16h;
typedef __attribute__((ext_vector_type(8)))  _Float16 v8h;
typedef __attribute__((ext_vector_type(4)))  _Float16 v4h;
typedef __attribute__((ext_vector_type(8)))  float    v8f;

#define EPS 1e-5f
#define BSTRIDE 40   // LDS row stride in halfwords (80 B, 16B-aligned rows)

// ---------------------------------------------------------------------------
// Generic WMMA GEMM: C[m][n] = alpha * sum_k A[m][k] * B[k][n] (+ bias[m])
//   ACONTIG==1 : a_cs==1, A fragment via 4x global_load_b128
//   BK==1      : b_rs==1 (k contiguous), staging via b128 + ds_store_b64
//   BK==0      : b_cs==1 (n contiguous), staging via b128 + 4x ds_store_b16
// Block = 128 threads (4 waves); block C tile = 64(M) x 32(N): each wave owns
// one 16-row strip and two 16x16 N-tiles sharing its A fragment (2 WMMA/chunk).
// M, N multiples of 16 (all call sites). N tail (N%32) handled by column
// clamping in staging + block-uniform store guard.
// ---------------------------------------------------------------------------
template <int ACONTIG, int BK>
__global__ __launch_bounds__(128)
void csam_wmma_gemm(const float* __restrict__ A,
                    const float* __restrict__ Bm,
                    const float* __restrict__ bias,
                    float* __restrict__ C,
                    int M, int N, int K,
                    int a_rs, int a_cs, int b_rs, int b_cs, int c_rs,
                    long long aOuter, long long aInner,
                    long long bOuter, long long bInner,
                    long long cOuter, long long cInner,
                    int innerCount, float alpha)
{
    __shared__ __align__(16) _Float16 sB[32 * BSTRIDE];

    const int z  = blockIdx.z;
    const int zb = z / innerCount;
    const int zi = z - zb * innerCount;
    A  += zb * aOuter + zi * aInner;
    Bm += zb * bOuter + zi * bInner;
    C  += zb * cOuter + zi * cInner;

    const int wave = threadIdx.x >> 5;
    const int lane = threadIdx.x & 31;
    const int tileM  = blockIdx.y * 4 + wave;
    const bool active = (tileM * 16) < M;        // wave-uniform
    const int row0 = tileM * 16;
    const int col0 = blockIdx.x * 32;

    const int lrow = lane & 15;                  // row (A) / col (B,C) in tile
    const int lhi  = lane >> 4;                  // lane-half selector

    const int arow = min(row0 + lrow, M - 1);    // clamp for inactive waves
    const float* Ap = A + (long long)arow * a_rs;

    v8f acc0 = {0.f, 0.f, 0.f, 0.f, 0.f, 0.f, 0.f, 0.f};
    v8f acc1 = {0.f, 0.f, 0.f, 0.f, 0.f, 0.f, 0.f, 0.f};

    const int kfull = K & ~31;

    // ===================== full 32-wide K chunks (no masking) =============
    for (int k0 = 0; k0 < kfull; k0 += 32) {
        __syncthreads();
        if (BK) {
            // k contiguous in memory: one float4 + one packed ds_store_b64
#pragma unroll
            for (int i = 0; i < 2; ++i) {
                const int lin = (int)threadIdx.x + 128 * i;    // 0..255
                const int n = lin >> 3;                        // 0..31
                const int k = (lin & 7) * 4;                   // 0..28
                const float4 p = *(const float4*)
                    (Bm + (long long)min(col0 + n, N - 1) * b_cs + k0 + k);
                v4h h; h[0] = (_Float16)p.x; h[1] = (_Float16)p.y;
                       h[2] = (_Float16)p.z; h[3] = (_Float16)p.w;
                *(v4h*)&sB[n * BSTRIDE + k] = h;
            }
        } else {
            // n contiguous in memory: one float4 + 4 transposed ds_store_b16
#pragma unroll
            for (int i = 0; i < 2; ++i) {
                const int lin = (int)threadIdx.x + 128 * i;
                const int k = lin >> 3;                        // 0..31
                const int n = (lin & 7) * 4;                   // 0..28
                const float4 p = *(const float4*)
                    (Bm + (long long)(k0 + k) * b_rs + min(col0 + n, N - 4));
                sB[(n + 0) * BSTRIDE + k] = (_Float16)p.x;
                sB[(n + 1) * BSTRIDE + k] = (_Float16)p.y;
                sB[(n + 2) * BSTRIDE + k] = (_Float16)p.z;
                sB[(n + 3) * BSTRIDE + k] = (_Float16)p.w;
            }
        }
        __syncthreads();

        // ---- A fragment (16x32 f16, ISA layout) ----
        v16h af;
        if (ACONTIG) {
            const float4* Alo = (const float4*)(Ap + k0 + lhi * 8);
            const float4* Ahi = (const float4*)(Ap + k0 + 16 + lhi * 8);
            const float4 a0 = Alo[0], a1 = Alo[1];
            const float4 a2 = Ahi[0], a3 = Ahi[1];
            af[0]  = (_Float16)a0.x; af[1]  = (_Float16)a0.y;
            af[2]  = (_Float16)a0.z; af[3]  = (_Float16)a0.w;
            af[4]  = (_Float16)a1.x; af[5]  = (_Float16)a1.y;
            af[6]  = (_Float16)a1.z; af[7]  = (_Float16)a1.w;
            af[8]  = (_Float16)a2.x; af[9]  = (_Float16)a2.y;
            af[10] = (_Float16)a2.z; af[11] = (_Float16)a2.w;
            af[12] = (_Float16)a3.x; af[13] = (_Float16)a3.y;
            af[14] = (_Float16)a3.z; af[15] = (_Float16)a3.w;
        } else {
#pragma unroll
            for (int v = 0; v < 8; ++v) {
                const int kk = (v < 4) ? (lhi * 8 + 2 * v)
                                       : (16 + lhi * 8 + 2 * (v - 4));
                af[2 * v]     = (_Float16)Ap[(long long)(k0 + kk) * a_cs];
                af[2 * v + 1] = (_Float16)Ap[(long long)(k0 + kk + 1) * a_cs];
            }
        }

        // ---- two B fragments from LDS, two WMMAs sharing af ----
        {
            const v8h b0 = *(const v8h*)&sB[lrow * BSTRIDE + lhi * 8];
            const v8h b1 = *(const v8h*)&sB[lrow * BSTRIDE + 16 + lhi * 8];
            const v16h bf = __builtin_shufflevector(
                b0, b1, 0, 1, 2, 3, 4, 5, 6, 7, 8, 9, 10, 11, 12, 13, 14, 15);
            acc0 = __builtin_amdgcn_wmma_f32_16x16x32_f16(
                       false, af, false, bf, (short)0, acc0, false, false);
        }
        {
            const v8h b0 = *(const v8h*)&sB[(16 + lrow) * BSTRIDE + lhi * 8];
            const v8h b1 = *(const v8h*)&sB[(16 + lrow) * BSTRIDE + 16 + lhi * 8];
            const v16h bf = __builtin_shufflevector(
                b0, b1, 0, 1, 2, 3, 4, 5, 6, 7, 8, 9, 10, 11, 12, 13, 14, 15);
            acc1 = __builtin_amdgcn_wmma_f32_16x16x32_f16(
                       false, af, false, bf, (short)0, acc1, false, false);
        }
    }

    // ===================== masked K tail (K % 32 != 0) =====================
    if (kfull < K) {
        const int k0 = kfull;
        __syncthreads();
        // generic scalar masked staging: 32n x 32k, 8 elements per thread
#pragma unroll
        for (int i = 0; i < 8; ++i) {
            const int lin = (int)threadIdx.x + 128 * i;        // 0..1023
            const int n = lin >> 5;                            // 0..31
            const int k = lin & 31;                            // 0..31
            const int kk = k0 + k;
            const int kc = min(kk, K - 1);
            const int nc = min(col0 + n, N - 1);
            const float v = Bm[(long long)kc * b_rs + (long long)nc * b_cs];
            sB[n * BSTRIDE + k] = (_Float16)(kk < K ? v : 0.f);
        }
        __syncthreads();

        v16h af;
#pragma unroll
        for (int v = 0; v < 8; ++v) {
            const int kk = (v < 4) ? (lhi * 8 + 2 * v)
                                   : (16 + lhi * 8 + 2 * (v - 4));
            const int k1 = k0 + kk, k2 = k1 + 1;
            const float x1 = Ap[(long long)min(k1, K - 1) * a_cs];
            const float x2 = Ap[(long long)min(k2, K - 1) * a_cs];
            af[2 * v]     = (_Float16)(k1 < K ? x1 : 0.f);
            af[2 * v + 1] = (_Float16)(k2 < K ? x2 : 0.f);
        }
        {
            const v8h b0 = *(const v8h*)&sB[lrow * BSTRIDE + lhi * 8];
            const v8h b1 = *(const v8h*)&sB[lrow * BSTRIDE + 16 + lhi * 8];
            const v16h bf = __builtin_shufflevector(
                b0, b1, 0, 1, 2, 3, 4, 5, 6, 7, 8, 9, 10, 11, 12, 13, 14, 15);
            acc0 = __builtin_amdgcn_wmma_f32_16x16x32_f16(
                       false, af, false, bf, (short)0, acc0, false, false);
        }
        {
            const v8h b0 = *(const v8h*)&sB[(16 + lrow) * BSTRIDE + lhi * 8];
            const v8h b1 = *(const v8h*)&sB[(16 + lrow) * BSTRIDE + 16 + lhi * 8];
            const v16h bf = __builtin_shufflevector(
                b0, b1, 0, 1, 2, 3, 4, 5, 6, 7, 8, 9, 10, 11, 12, 13, 14, 15);
            acc1 = __builtin_amdgcn_wmma_f32_16x16x32_f16(
                       false, af, false, bf, (short)0, acc1, false, false);
        }
    }

    // ===================== store ==========================================
    if (active) {
        const int cn0 = col0 + lrow;
#pragma unroll
        for (int r = 0; r < 8; ++r) {
            const int cm = row0 + r + 8 * lhi;
            float v = acc0[r] * alpha;
            if (bias) v += bias[cm];
            C[(long long)cm * c_rs + cn0] = v;
        }
        if (col0 + 16 < N) {                     // block-uniform guard
            const int cn1 = col0 + 16 + lrow;
#pragma unroll
            for (int r = 0; r < 8; ++r) {
                const int cm = row0 + r + 8 * lhi;
                float v = acc1[r] * alpha;
                if (bias) v += bias[cm];
                C[(long long)cm * c_rs + cn1] = v;
            }
        }
    }
}

// ---------------------------------------------------------------------------
// Row-wise softmax, one wave32 per row (4 rows per 128-thread block).
// ---------------------------------------------------------------------------
__global__ __launch_bounds__(128)
void csam_softmax_rows(float* __restrict__ data, int cols)
{
    const long long row = (long long)blockIdx.x * 4 + (threadIdx.x >> 5);
    const int lane = threadIdx.x & 31;
    float* p = data + row * (long long)cols;

    float mx = -3.402823466e38f;
    for (int i = lane; i < cols; i += 32) mx = fmaxf(mx, p[i]);
#pragma unroll
    for (int o = 16; o > 0; o >>= 1) mx = fmaxf(mx, __shfl_xor(mx, o, 32));

    float sum = 0.f;
    for (int i = lane; i < cols; i += 32) {
        const float e = __expf(p[i] - mx);
        p[i] = e;
        sum += e;
    }
#pragma unroll
    for (int o = 16; o > 0; o >>= 1) sum += __shfl_xor(sum, o, 32);

    const float inv = 1.f / sum;
    for (int i = lane; i < cols; i += 32) p[i] *= inv;
}

// ---------------------------------------------------------------------------
// Bilinear upsample (half-pixel centers, edge clamp), NCHW, fp32.
// ---------------------------------------------------------------------------
__global__ void csam_upsample_bilinear(const float* __restrict__ in,
                                       float* __restrict__ out,
                                       int CH, int Hi, int Wi, int Ho, int Wo)
{
    const long long idx = (long long)blockIdx.x * blockDim.x + threadIdx.x;
    const long long total = (long long)CH * Ho * Wo;
    if (idx >= total) return;

    const int wo = (int)(idx % Wo);
    const int ho = (int)((idx / Wo) % Ho);
    const long long c = idx / ((long long)Wo * Ho);

    const float sy = (float)Hi / (float)Ho;
    const float sx = (float)Wi / (float)Wo;
    const float fy = ((float)ho + 0.5f) * sy - 0.5f;
    const float fx = ((float)wo + 0.5f) * sx - 0.5f;

    int y0 = (int)floorf(fy);
    int x0 = (int)floorf(fx);
    const float wy = fy - (float)y0;
    const float wx = fx - (float)x0;
    int y1 = y0 + 1, x1 = x0 + 1;
    y0 = min(max(y0, 0), Hi - 1);
    y1 = min(max(y1, 0), Hi - 1);
    x0 = min(max(x0, 0), Wi - 1);
    x1 = min(max(x1, 0), Wi - 1);

    const float* pc = in + c * (long long)Hi * Wi;
    const float v00 = pc[y0 * Wi + x0];
    const float v01 = pc[y0 * Wi + x1];
    const float v10 = pc[y1 * Wi + x0];
    const float v11 = pc[y1 * Wi + x1];
    const float top = v00 + (v01 - v00) * wx;
    const float bot = v10 + (v11 - v10) * wx;
    out[idx] = top + (bot - top) * wy;
}

// ---------------------------------------------------------------------------
// y = residual + silu(batchnorm(conv_out)), NCHW fp32.
// ---------------------------------------------------------------------------
__global__ void csam_bn_silu_res(const float* __restrict__ conv,
                                 const float* __restrict__ res,
                                 float* __restrict__ out,
                                 const float* __restrict__ g,
                                 const float* __restrict__ b,
                                 const float* __restrict__ m,
                                 const float* __restrict__ v,
                                 int C, int HW, long long total)
{
    const long long i = (long long)blockIdx.x * blockDim.x + threadIdx.x;
    if (i >= total) return;
    const int c = (int)((i / HW) % C);
    const float s = g[c] * rsqrtf(v[c] + EPS);
    const float y = (conv[i] - m[c]) * s + b[c];
    out[i] = res[i] + y / (1.f + __expf(-y));   // y * sigmoid(y)
}

// ---------------------------------------------------------------------------
// Host-side launch helper (dispatches on A contiguity / B orientation)
// ---------------------------------------------------------------------------
static inline void gemm(hipStream_t s,
                        const float* A, const float* B, const float* bias, float* C,
                        int M, int N, int K,
                        int a_rs, int a_cs, int b_rs, int b_cs, int c_rs,
                        long long aO, long long aI, long long bO, long long bI,
                        long long cO, long long cI, int inner, int zcount,
                        float alpha)
{
    dim3 grid((N + 31) / 32, (M + 63) / 64, zcount);
    const dim3 blk(128);
    if (a_cs == 1) {
        if (b_rs == 1)
            csam_wmma_gemm<1, 1><<<grid, blk, 0, s>>>(A, B, bias, C, M, N, K,
                a_rs, a_cs, b_rs, b_cs, c_rs, aO, aI, bO, bI, cO, cI, inner, alpha);
        else
            csam_wmma_gemm<1, 0><<<grid, blk, 0, s>>>(A, B, bias, C, M, N, K,
                a_rs, a_cs, b_rs, b_cs, c_rs, aO, aI, bO, bI, cO, cI, inner, alpha);
    } else {
        if (b_rs == 1)
            csam_wmma_gemm<0, 1><<<grid, blk, 0, s>>>(A, B, bias, C, M, N, K,
                a_rs, a_cs, b_rs, b_cs, c_rs, aO, aI, bO, bI, cO, cI, inner, alpha);
        else
            csam_wmma_gemm<0, 0><<<grid, blk, 0, s>>>(A, B, bias, C, M, N, K,
                a_rs, a_cs, b_rs, b_cs, c_rs, aO, aI, bO, bI, cO, cI, inner, alpha);
    }
}

extern "C" void kernel_launch(void* const* d_in, const int* in_sizes, int n_in,
                              void* d_out, int out_size, void* d_ws, size_t ws_size,
                              hipStream_t stream)
{
    (void)in_sizes; (void)n_in; (void)out_size; (void)ws_size;

    // ---- inputs (setup_inputs order) ----
    const float* x0   = (const float*)d_in[0];   // (2,256,40,40)
    const float* x1   = (const float*)d_in[1];   // (2,512,20,20)
    const float* x2   = (const float*)d_in[2];   // (2,1024,10,10)
    const float* wq1  = (const float*)d_in[3];   // (256,512)
    const float* wkv1 = (const float*)d_in[4];   // (256,512)
    const float* wkv2 = (const float*)d_in[5];   // (256,1024)
    const float* ipw  = (const float*)d_in[6];   // (768,256)
    const float* ipb  = (const float*)d_in[7];   // (768,)
    const float* outw = (const float*)d_in[8];   // (256,256)
    const float* outb = (const float*)d_in[9];   // (256,)
    const float* f0w  = (const float*)d_in[10];  // (256,512)
    const float* f1w  = (const float*)d_in[11];  // (512,256)
    const float* bn0g = (const float*)d_in[12];
    const float* bn0b = (const float*)d_in[13];
    const float* bn0m = (const float*)d_in[14];
    const float* bn0v = (const float*)d_in[15];
    const float* bn1g = (const float*)d_in[16];
    const float* bn1b = (const float*)d_in[17];
    const float* bn1m = (const float*)d_in[18];
    const float* bn1v = (const float*)d_in[19];

    const int Bn = 2, E = 256, NH = 8, HD = 32;
    const int N1 = 400, N0 = 1600;               // 20x20 and 40x40
    const float scale = 0.17677669529663687f;     // 1/sqrt(32)

    float* WS  = (float*)d_ws;
    float* OUT = (float*)d_out;
    float* outY0 = OUT;                           // 2*256*1600 = 819200
    float* outY1 = OUT + 819200;                  // 2*512*400  = 409600
    float* outX2 = OUT + 819200 + 409600;         // 2*1024*100 = 204800

    const float* wq = ipw;                        // rows   0..255
    const float* wk = ipw + (long long)E * E;     // rows 256..511
    const float* wv = ipw + 2LL * E * E;          // rows 512..767
    const float* bq = ipb, *bk = ipb + E, *bv = ipb + 2 * E;

    // =================== Phase A workspace (floats) ===================
    float* UP20 = WS;                  // (2,1024,400)  819200
    float* Q1T  = WS + 819200;         // (2,256,400)   204800
    float* KV20 = WS + 1024000;        // (2,256,400)
    float* QPT  = WS + 1228800;        // (2,256,400)
    float* KPT  = WS + 1433600;        // (2,256,400)
    float* VPT  = WS + 1638400;        // (2,256,400)
    float* SC1  = WS + 1843200;        // (2,8,400,400) 2560000
    float* CTX1 = WS + 4403200;        // (2,256,400)
    float* O1   = WS + 4608000;        // (2,256,400)
    float* CF1  = WS + 4812800;        // (2,512,400)   409600

    // ---- Stage 1: attention at 20x20 ----
    {
        long long tot = 2LL * 1024 * 400;
        csam_upsample_bilinear<<<dim3((unsigned)((tot + 255) / 256)), dim3(256), 0, stream>>>(
            x2, UP20, 2 * 1024, 10, 10, 20, 20);
    }
    // q1 = wq1 @ x1          (E x 512) x (512 x 400)
    gemm(stream, wq1, x1, nullptr, Q1T, E, N1, 512, 512, 1, N1, 1, N1,
         0, 0, 512LL * N1, 0, (long long)E * N1, 0, 1, Bn, 1.f);
    // kv = wkv2 @ up20       (E x 1024) x (1024 x 400)
    gemm(stream, wkv2, UP20, nullptr, KV20, E, N1, 1024, 1024, 1, N1, 1, N1,
         0, 0, 1024LL * N1, 0, (long long)E * N1, 0, 1, Bn, 1.f);
    // q/k/v projections (kept in (E, N) layout)
    gemm(stream, wq, Q1T,  bq, QPT, E, N1, E, E, 1, N1, 1, N1,
         0, 0, (long long)E * N1, 0, (long long)E * N1, 0, 1, Bn, 1.f);
    gemm(stream, wk, KV20, bk, KPT, E, N1, E, E, 1, N1, 1, N1,
         0, 0, (long long)E * N1, 0, (long long)E * N1, 0, 1, Bn, 1.f);
    gemm(stream, wv, KV20, bv, VPT, E, N1, E, E, 1, N1, 1, N1,
         0, 0, (long long)E * N1, 0, (long long)E * N1, 0, 1, Bn, 1.f);
    // scores[b,h] = scale * qp^T kp : M=N=400, K=32, batched over b*h=16
    gemm(stream, QPT, KPT, nullptr, SC1, N1, N1, HD,
         1, N1, N1, 1, N1,
         (long long)E * N1, (long long)HD * N1,
         (long long)E * N1, (long long)HD * N1,
         8LL * N1 * N1, (long long)N1 * N1, NH, Bn * NH, scale);
    csam_softmax_rows<<<dim3(Bn * NH * N1 / 4), dim3(128), 0, stream>>>(SC1, N1);
    // ctx[b, h*32+d, q] = sum_k vp[d,k] * attn[q,k] : M=32, N=400, K=400
    gemm(stream, VPT, SC1, nullptr, CTX1, HD, N1, N1,
         N1, 1, 1, N1, N1,
         (long long)E * N1, (long long)HD * N1,
         8LL * N1 * N1, (long long)N1 * N1,
         (long long)E * N1, (long long)HD * N1, NH, Bn * NH, 1.f);
    // out projection
    gemm(stream, outw, CTX1, outb, O1, E, N1, E, E, 1, N1, 1, N1,
         0, 0, (long long)E * N1, 0, (long long)E * N1, 0, 1, Bn, 1.f);
    // fuse1 conv: (512 x 256) x (256 x 400)
    gemm(stream, f1w, O1, nullptr, CF1, 512, N1, E, E, 1, N1, 1, N1,
         0, 0, (long long)E * N1, 0, 512LL * N1, 0, 1, Bn, 1.f);
    // y1 = x1 + silu(bn1(CF1))  -> d_out
    {
        long long tot = 2LL * 512 * N1;
        csam_bn_silu_res<<<dim3((unsigned)((tot + 255) / 256)), dim3(256), 0, stream>>>(
            CF1, x1, outY1, bn1g, bn1b, bn1m, bn1v, 512, N1, tot);
    }
    // passthrough x2 -> d_out
    hipMemcpyAsync(outX2, x2, 204800 * sizeof(float), hipMemcpyDeviceToDevice, stream);

    // ============ Phase B workspace (reuses Phase A region) ============
    float* UPY1  = WS;                   // (2,512,1600)  1638400
    float* UP240 = WS + 1638400;         // (2,1024,1600) 3276800
    float* KVA   = WS + 4915200;         // (2,256,1600)  819200
    float* KVB   = WS + 5734400;
    float* QP0T  = WS + 6553600;
    float* KPA   = WS + 7372800;
    float* VPA   = WS + 8192000;
    float* KPB   = WS + 9011200;
    float* VPB   = WS + 9830400;
    float* SC    = WS + 10649600;        // (1600,1600)   2560000 (reused per b,h)
    float* CTX   = WS + 13209600;        // (2,256,1600)  819200  (reused per attn)
    float* FUSED = WS + 14028800;        // (2,512,1600)  1638400
    float* CF0   = WS + 15667200;        // (2,256,1600)  819200

    {
        long long tot = 2LL * 512 * N0;
        csam_upsample_bilinear<<<dim3((unsigned)((tot + 255) / 256)), dim3(256), 0, stream>>>(
            outY1, UPY1, 2 * 512, 20, 20, 40, 40);
        tot = 2LL * 1024 * N0;
        csam_upsample_bilinear<<<dim3((unsigned)((tot + 255) / 256)), dim3(256), 0, stream>>>(
            x2, UP240, 2 * 1024, 10, 10, 40, 40);
    }
    // kvA = wkv1 @ up(y1), kvB = wkv2 @ up(x2)
    gemm(stream, wkv1, UPY1, nullptr, KVA, E, N0, 512, 512, 1, N0, 1, N0,
         0, 0, 512LL * N0, 0, (long long)E * N0, 0, 1, Bn, 1.f);
    gemm(stream, wkv2, UP240, nullptr, KVB, E, N0, 1024, 1024, 1, N0, 1, N0,
         0, 0, 1024LL * N0, 0, (long long)E * N0, 0, 1, Bn, 1.f);
    // shared query projection of x0
    gemm(stream, wq, x0, bq, QP0T, E, N0, E, E, 1, N0, 1, N0,
         0, 0, (long long)E * N0, 0, (long long)E * N0, 0, 1, Bn, 1.f);
    // k/v projections for both attention branches
    gemm(stream, wk, KVA, bk, KPA, E, N0, E, E, 1, N0, 1, N0,
         0, 0, (long long)E * N0, 0, (long long)E * N0, 0, 1, Bn, 1.f);
    gemm(stream, wv, KVA, bv, VPA, E, N0, E, E, 1, N0, 1, N0,
         0, 0, (long long)E * N0, 0, (long long)E * N0, 0, 1, Bn, 1.f);
    gemm(stream, wk, KVB, bk, KPB, E, N0, E, E, 1, N0, 1, N0,
         0, 0, (long long)E * N0, 0, (long long)E * N0, 0, 1, Bn, 1.f);
    gemm(stream, wv, KVB, bv, VPB, E, N0, E, E, 1, N0, 1, N0,
         0, 0, (long long)E * N0, 0, (long long)E * N0, 0, 1, Bn, 1.f);

    // Two attention branches over x0; oA -> fused channels [0,256),
    // oB -> fused channels [256,512). Score buffer reused per (b,h).
    for (int branch = 0; branch < 2; ++branch) {
        const float* KP = branch ? KPB : KPA;
        const float* VP = branch ? VPB : VPA;
        for (int b = 0; b < Bn; ++b) {
            for (int h = 0; h < NH; ++h) {
                const long long off = (long long)b * E * N0 + (long long)h * HD * N0;
                // scores: M=N=1600, K=32
                gemm(stream, QP0T + off, KP + off, nullptr, SC, N0, N0, HD,
                     1, N0, N0, 1, N0, 0, 0, 0, 0, 0, 0, 1, 1, scale);
                csam_softmax_rows<<<dim3(N0 / 4), dim3(128), 0, stream>>>(SC, N0);
                // ctx: M=32, N=1600, K=1600
                gemm(stream, VP + off, SC, nullptr, CTX + off, HD, N0, N0,
                     N0, 1, 1, N0, N0, 0, 0, 0, 0, 0, 0, 1, 1, 1.f);
            }
        }
        // out projection -> fused half (concat for free)
        float* Cdst = FUSED + (long long)branch * E * N0;
        gemm(stream, outw, CTX, outb, Cdst, E, N0, E, E, 1, N0, 1, N0,
             0, 0, (long long)E * N0, 0, 512LL * N0, 0, 1, Bn, 1.f);
    }

    // fuse0 conv: (256 x 512) x (512 x 1600)
    gemm(stream, f0w, FUSED, nullptr, CF0, E, N0, 512, 512, 1, N0, 1, N0,
         0, 0, 512LL * N0, 0, (long long)E * N0, 0, 1, Bn, 1.f);
    // y0 = x0 + silu(bn0(CF0)) -> d_out
    {
        long long tot = 2LL * 256 * N0;
        csam_bn_silu_res<<<dim3((unsigned)((tot + 255) / 256)), dim3(256), 0, stream>>>(
            CF0, x0, outY0, bn0g, bn0b, bn0m, bn0v, 256, N0, tot);
    }
}
